// interpretableMultiHeadAttention_58119497449867
// MI455X (gfx1250) — matrix operand order repaired and checked by reading, not compiled, not run on hardware
//
#include <hip/hip_runtime.h>
#include <hip/hip_bf16.h>

// ---------------------------------------------------------------------------
// interpretable multi-head attention for MI455X (gfx1250), bf16 WMMA pipeline
//   B=4, S=2048, D=1024, H=4, DH=256
// ---------------------------------------------------------------------------
#define B_  4
#define S_  2048
#define D_  1024
#define H_  4
#define DH_ 256

typedef __attribute__((ext_vector_type(16))) __bf16 v16bf;
typedef __attribute__((ext_vector_type(8)))  __bf16 v8bf;
typedef __attribute__((ext_vector_type(4)))  __bf16 v4bf;
typedef __attribute__((ext_vector_type(8)))  float  v8f;
typedef __attribute__((ext_vector_type(4)))  float  v4f;

// ---- CDNA5 async global->LDS copy (ISA 10: GLOBAL_LOAD_ASYNC_TO_LDS) -------
__device__ __forceinline__ void async_b128(void* lds_dst, const void* gsrc) {
  const uint32_t lo = (uint32_t)(uintptr_t)lds_dst;  // low 32b = LDS offset
  asm volatile("global_load_async_to_lds_b128 %0, %1, off"
               :
               : "v"(lo), "v"((uint64_t)(uintptr_t)gsrc)
               : "memory");
}
__device__ __forceinline__ void wait_async0() {
  asm volatile("s_wait_asynccnt 0x0" ::: "memory");
}

// ---- WMMA fragment helpers (wave32; contiguous k = 16*(lane/16) + e in A and B) ----------

// A fragment: 16x32 bf16 tile, source row-major p[m*ld + k].
__device__ __forceinline__ v16bf frag_a_rm(const __bf16* p, int ld) {
  const int lane = threadIdx.x & 31;
  const int m = lane & 15, hf = lane >> 4;
  v16bf a;
#pragma unroll
  for (int e = 0; e < 16; ++e) {
    const int kk = hf * 16 + e;
    a[e] = p[m * ld + kk];
  }
  return a;
}

// B fragment: B[k][n], source stored N-major: p[n*ld + k].
// Contiguous 16 bf16 per half-lane -> lowers to 2x ds_load_b128.
__device__ __forceinline__ v16bf frag_b_nk(const __bf16* p, int ld) {
  const int lane = threadIdx.x & 31;
  const int n = lane & 15, hf = lane >> 4;
  v16bf b;
#pragma unroll
  for (int e = 0; e < 16; ++e) b[e] = p[n * ld + hf * 16 + e];
  return b;
}

__device__ __forceinline__ v8f wmma_bf16(v16bf a, v16bf b, v8f c) {
  return __builtin_amdgcn_wmma_f32_16x16x32_bf16(false, a, false, b,
                                                 (short)0, c, false, false);
}

// ---------------------------------------------------------------------------
// Kernel 1: projections.  job 0..3 -> qs head, 4..7 -> ks head, 8 -> vs.
// GEMM [8192 x 1024] * W^T[1024 x 256] + bias, bf16 out.
// qs/ks stored [B,H,S,DH]; vs stored TRANSPOSED [B,DH,S] so the PV matmul
// can async-stage V with key-contiguous rows and use b128 fragment loads.
// ---------------------------------------------------------------------------
__global__ __launch_bounds__(256) void proj_kernel(
    const float* __restrict__ q, const float* __restrict__ k,
    const float* __restrict__ v, const float* __restrict__ qW,
    const float* __restrict__ kW, const float* __restrict__ vW,
    const float* __restrict__ qb, const float* __restrict__ kb,
    const float* __restrict__ vb, __bf16* __restrict__ qs,
    __bf16* __restrict__ ks, __bf16* __restrict__ vs) {
  __shared__ __bf16 at[128 * 40];  // A tile, ld=40 (pad)
  __shared__ __bf16 bt[64 * 40];   // W tile (N-major), ld=40

  const int job = blockIdx.z;
  const int m0 = blockIdx.x * 128;
  const int n0 = blockIdx.y * 64;

  const float *X, *W, *bias;
  __bf16* dstP = vs;
  size_t strideB = 0;
  if (job < 4) {
    X = q; W = qW + (size_t)job * DH_ * D_; bias = qb + job * DH_;
    dstP = qs + (size_t)job * S_ * DH_; strideB = (size_t)H_ * S_ * DH_;
  } else if (job < 8) {
    X = k; W = kW + (size_t)(job - 4) * DH_ * D_; bias = kb + (job - 4) * DH_;
    dstP = ks + (size_t)(job - 4) * S_ * DH_; strideB = (size_t)H_ * S_ * DH_;
  } else {
    X = v; W = vW; bias = vb;
  }

  const int tid = threadIdx.x;
  const int w  = tid >> 5;
  const int wm = (w >> 1) * 32;
  const int wn = (w & 1) * 32;

  v8f acc[2][2] = {};

  for (int k0 = 0; k0 < D_; k0 += 32) {
    __syncthreads();
    // stage A: 128x32 fp32 -> bf16 (float4 + packed cvt, 4 quads/thread)
#pragma unroll
    for (int i = 0; i < 4; ++i) {
      const int idx = i * 256 + tid;
      const int r = idx >> 3, c4 = (idx & 7) << 2;
      const v4f xv = *(const v4f*)(X + (size_t)(m0 + r) * D_ + k0 + c4);
      *(v4bf*)(at + r * 40 + c4) = __builtin_convertvector(xv, v4bf);
    }
    // stage W: 64(n) x 32(k) N-major (2 quads/thread)
#pragma unroll
    for (int i = 0; i < 2; ++i) {
      const int idx = i * 256 + tid;
      const int r = idx >> 3, c4 = (idx & 7) << 2;
      const v4f wv = *(const v4f*)(W + (size_t)(n0 + r) * D_ + k0 + c4);
      *(v4bf*)(bt + r * 40 + c4) = __builtin_convertvector(wv, v4bf);
    }
    if (k0 + 32 < D_) {  // prefetch next A K-slab (128 rows x 128B)
      const int pr = tid >> 1, pc = (tid & 1) * 64;
      __builtin_prefetch((const char*)(X + (size_t)(m0 + pr) * D_ + k0 + 32) +
                             pc, 0, 1);
    }
    __syncthreads();
#pragma unroll
    for (int i = 0; i < 2; ++i) {
      const v16bf a = frag_a_rm(at + (wm + 16 * i) * 40, 40);
#pragma unroll
      for (int j = 0; j < 2; ++j) {
        const v16bf b = frag_b_nk(bt + (wn + 16 * j) * 40, 40);
        acc[i][j] = wmma_bf16(a, b, acc[i][j]);
      }
    }
  }

  // epilogue: +bias, bf16 store (uniform branch on job; EXEC stays full)
  const int lane = tid & 31, cn = lane & 15, ch = lane >> 4;
  if (job < 8) {
#pragma unroll
    for (int i = 0; i < 2; ++i)
#pragma unroll
      for (int j = 0; j < 2; ++j)
#pragma unroll
        for (int r = 0; r < 8; ++r) {
          const int m = m0 + wm + 16 * i + r + 8 * ch;
          const int n = n0 + wn + 16 * j + cn;
          const int bb = m >> 11, s = m & (S_ - 1);
          dstP[(size_t)bb * strideB + (size_t)s * DH_ + n] =
              (__bf16)(acc[i][j][r] + bias[n]);
        }
  } else {  // vs: transposed [B, DH, S]
#pragma unroll
    for (int i = 0; i < 2; ++i)
#pragma unroll
      for (int j = 0; j < 2; ++j)
#pragma unroll
        for (int r = 0; r < 8; ++r) {
          const int m = m0 + wm + 16 * i + r + 8 * ch;
          const int n = n0 + wn + 16 * j + cn;
          const int bb = m >> 11, s = m & (S_ - 1);
          dstP[((size_t)bb * DH_ + n) * S_ + s] =
              (__bf16)(acc[i][j][r] + bias[n]);
        }
  }
}

// ---------------------------------------------------------------------------
// Kernel 2: attention. One WG = (b, h, 16 query rows). Full 16x2048 f32 score
// block in LDS (CDNA5 320KB WGP LDS). K/V/Q staged with
// global_load_async_to_lds_b128 (ASYNCcnt). Dynamic LDS = 276096 B.
// ---------------------------------------------------------------------------
__global__ __launch_bounds__(256) void attn_kernel(
    const __bf16* __restrict__ qs, const __bf16* __restrict__ ksm,
    const __bf16* __restrict__ vsm, float* __restrict__ attn,
    __bf16* __restrict__ head) {
  extern __shared__ char smem[];
  float*  scores = (float*)smem;                     // 16 x 2048 f32
  __bf16* pbuf   = (__bf16*)(smem + 16 * 2048 * 4);  // 16 x 2056 bf16
  __bf16* kvbuf  = pbuf + 16 * 2056;   // phase A: 128x264; phase C: 256x136
  __bf16* qtile  = kvbuf + 256 * 136;                // 16 x 264 bf16
  float*  red    = (float*)(qtile + 16 * 264);       // 256 f32
  float*  rowm   = red + 256;                        // 16
  float*  rowi   = rowm + 16;                        // 16

  const int tid = threadIdx.x;
  const int b = blockIdx.z, h = blockIdx.y;
  const int s0 = blockIdx.x * 16;
  const int w = tid >> 5;
  const int lane = tid & 31, cn = lane & 15, ch = lane >> 4;

  const __bf16* qbase = qs + (((size_t)b * H_ + h) * S_ + s0) * DH_;
  const __bf16* kbase = ksm + (((size_t)b * H_ + h) * S_) * DH_;
  const __bf16* vbase = vsm + (size_t)b * DH_ * S_;  // [DH][S] per batch

  // stage Q tile 16x256 bf16 via async DMA (2 x b128 per thread)
#pragma unroll
  for (int i = 0; i < 2; ++i) {
    const int idx = i * 256 + tid;
    const int r = idx >> 5, c = idx & 31;
    async_b128((uint4*)qtile + r * 33 + c, (const uint4*)qbase + r * 32 + c);
  }

  // ---- Phase A: scores = (Q K^T) / sqrt(DH) ----
  for (int kb = 0; kb < 16; ++kb) {
    __syncthreads();
    {  // async-stage 128-key block (128 x 256 bf16 = 64KB, 16 b128/thread)
      const uint4* src = (const uint4*)(kbase + (size_t)(kb * 128) * DH_);
#pragma unroll
      for (int i = 0; i < 16; ++i) {
        const int idx = i * 256 + tid;
        const int r = idx >> 5, c = idx & 31;
        async_b128((uint4*)kvbuf + r * 33 + c, src + r * 32 + c);
      }
    }
    wait_async0();
    __syncthreads();
    // wave w handles keys [kb*128 + w*16, +16); K-dim 256 in 8 WMMA steps
    v8f sacc = {};
#pragma unroll
    for (int ke = 0; ke < 8; ++ke) {
      const v16bf a = frag_a_rm(qtile + ke * 32, 264);
      const v16bf bf = frag_b_nk(kvbuf + (w * 16) * 264 + ke * 32, 264);
      sacc = wmma_bf16(a, bf, sacc);
    }
#pragma unroll
    for (int r = 0; r < 8; ++r) {
      const int m = r + 8 * ch;
      scores[m * 2048 + kb * 128 + w * 16 + cn] = sacc[r] * 0.0625f;
    }
  }
  __syncthreads();

  // ---- Phase B: softmax (16 threads/row, contiguous 128-elem chunks) ----
  const int row = tid >> 4, sub = tid & 15;
  float* srow = scores + row * 2048 + sub * 128;
  float mx = -3.0e38f;
#pragma unroll 8
  for (int i = 0; i < 32; ++i) {
    const v4f x = *(const v4f*)(srow + i * 4);
    mx = fmaxf(mx, fmaxf(fmaxf(x[0], x[1]), fmaxf(x[2], x[3])));
  }
  red[tid] = mx;
  __syncthreads();
  if (sub == 0) {
    float m2 = red[tid];
    for (int i = 1; i < 16; ++i) m2 = fmaxf(m2, red[tid + i]);
    rowm[row] = m2;
  }
  __syncthreads();
  const float rm = rowm[row];
  float sum = 0.f;
#pragma unroll 8
  for (int i = 0; i < 32; ++i) {
    v4f x = *(const v4f*)(srow + i * 4);
    x[0] = __expf(x[0] - rm); x[1] = __expf(x[1] - rm);
    x[2] = __expf(x[2] - rm); x[3] = __expf(x[3] - rm);
    *(v4f*)(srow + i * 4) = x;
    sum += (x[0] + x[1]) + (x[2] + x[3]);
  }
  red[tid] = sum;
  __syncthreads();
  if (sub == 0) {
    float s2 = 0.f;
    for (int i = 0; i < 16; ++i) s2 += red[tid + i];
    rowi[row] = 1.0f / s2;
  }
  __syncthreads();

  // normalize: float4 writes of attn probs + packed bf16 prob tile
  float* attn_base = attn + (((size_t)b * S_ + s0) * H_ + h) * (size_t)S_;
  for (int i = 0; i < 32; ++i) {  // 8192 quads / 256 threads
    const int idx = i * 256 + tid;
    const int r = idx >> 9, tq = (idx & 511) << 2;
    v4f p = *(const v4f*)(scores + r * 2048 + tq);
    p *= rowi[r];
    *(v4f*)(attn_base + (size_t)r * (H_ * S_) + tq) = p;
    *(v4bf*)(pbuf + r * 2056 + tq) = __builtin_convertvector(p, v4bf);
  }
  __syncthreads();

  // ---- Phase C: head = P @ V  (16x2048 @ 2048x256), V is [DH][S] ----
  v8f hacc[2] = {};
  const int n0 = w * 32;
  for (int kb = 0; kb < 16; ++kb) {
    __syncthreads();
    {  // async-stage V block transposed: 256 dims x 128 keys (uint4 pitch 17)
#pragma unroll
      for (int i = 0; i < 16; ++i) {
        const int idx = i * 256 + tid;
        const int r = idx >> 4, c = idx & 15;  // r = value-dim, c = key granule
        async_b128((uint4*)kvbuf + r * 17 + c,
                   (const uint4*)(vbase + (size_t)r * S_ + kb * 128) + c);
      }
    }
    wait_async0();
    __syncthreads();
#pragma unroll
    for (int ksb = 0; ksb < 4; ++ksb) {
      const v16bf a = frag_a_rm(pbuf + kb * 128 + ksb * 32, 2056);
#pragma unroll
      for (int j = 0; j < 2; ++j) {
        // B[k=key][n=dim] stored N-major as kvbuf[dim][key], ld=136
        const v16bf bf = frag_b_nk(kvbuf + (n0 + 16 * j) * 136 + ksb * 32, 136);
        hacc[j] = wmma_bf16(a, bf, hacc[j]);
      }
    }
  }

  // store head tile bf16 to [B,S,H,DH]
  __bf16* hbase = head + (((size_t)b * S_ + s0) * H_ + h) * DH_;
#pragma unroll
  for (int j = 0; j < 2; ++j)
#pragma unroll
    for (int r = 0; r < 8; ++r) {
      const int m = r + 8 * ch;
      hbase[(size_t)m * (H_ * DH_) + n0 + 16 * j + cn] = (__bf16)hacc[j][r];
    }
}

// ---------------------------------------------------------------------------
// Kernel 3: mean over heads fused into A-staging, then out = mean @ wH^T.
// GEMM [8192 x 256] * [256 x 1024], tile 64(M) x 128(N); whole K staged once.
// Dynamic LDS = (64+128)*264*2 = 101376 B.
// ---------------------------------------------------------------------------
__global__ __launch_bounds__(256) void outproj_kernel(
    const __bf16* __restrict__ head, const float* __restrict__ wH,
    float* __restrict__ out) {
  extern __shared__ char smem[];
  __bf16* atile = (__bf16*)smem;     // 64 x 264
  __bf16* btile = atile + 64 * 264;  // 128 x 264 (N-major wH tile)

  const int tid = threadIdx.x;
  const int m0 = blockIdx.x * 64;
  const int n0 = blockIdx.y * 128;

  // stage A = mean over heads, vectorized 8-wide bf16
  for (int i = 0; i < 8; ++i) {  // 2048 octets / 256 threads
    const int idx = i * 256 + tid;
    const int r = idx >> 5, c8 = (idx & 31) << 3;
    const int m = m0 + r, bb = m >> 11, s = m & (S_ - 1);
    const __bf16* hp = head + (((size_t)bb * S_ + s) * H_) * DH_ + c8;
    v8f sum = __builtin_convertvector(*(const v8bf*)hp, v8f) +
              __builtin_convertvector(*(const v8bf*)(hp + DH_), v8f) +
              __builtin_convertvector(*(const v8bf*)(hp + 2 * DH_), v8f) +
              __builtin_convertvector(*(const v8bf*)(hp + 3 * DH_), v8f);
    sum *= 0.25f;
    *(v8bf*)(atile + r * 264 + c8) = __builtin_convertvector(sum, v8bf);
  }
  // stage B: wH rows f in [n0, n0+128), all 256 K (float4 + packed cvt)
  for (int i = 0; i < 32; ++i) {  // 8192 quads / 256
    const int idx = i * 256 + tid;
    const int r = idx >> 6, c4 = (idx & 63) << 2;
    const v4f xv = *(const v4f*)(wH + (size_t)(n0 + r) * DH_ + c4);
    *(v4bf*)(btile + r * 264 + c4) = __builtin_convertvector(xv, v4bf);
  }
  __syncthreads();

  const int w = tid >> 5;
  const int wm = (w >> 2) * 32, wn = (w & 3) * 32;
  v8f acc[2][2] = {};
#pragma unroll
  for (int k0 = 0; k0 < 8; ++k0)
#pragma unroll
    for (int i = 0; i < 2; ++i) {
      const v16bf a = frag_a_rm(atile + (wm + 16 * i) * 264 + k0 * 32, 264);
#pragma unroll
      for (int j = 0; j < 2; ++j) {
        const v16bf b = frag_b_nk(btile + (wn + 16 * j) * 264 + k0 * 32, 264);
        acc[i][j] = wmma_bf16(a, b, acc[i][j]);
      }
    }

  const int lane = tid & 31, cn = lane & 15, ch = lane >> 4;
#pragma unroll
  for (int i = 0; i < 2; ++i)
#pragma unroll
    for (int j = 0; j < 2; ++j)
#pragma unroll
      for (int r = 0; r < 8; ++r) {
        const int m = m0 + wm + 16 * i + r + 8 * ch;
        const int n = n0 + wn + 16 * j + cn;
        out[(size_t)m * D_ + n] = acc[i][j][r];
      }
}

// ---------------------------------------------------------------------------
extern "C" void kernel_launch(void* const* d_in, const int* in_sizes, int n_in,
                              void* d_out, int out_size, void* d_ws,
                              size_t ws_size, hipStream_t stream) {
  (void)in_sizes; (void)n_in; (void)out_size; (void)ws_size;
  const float* q  = (const float*)d_in[0];
  const float* k  = (const float*)d_in[1];
  const float* v  = (const float*)d_in[2];
  const float* vW = (const float*)d_in[3];
  const float* vb = (const float*)d_in[4];
  const float* qW = (const float*)d_in[5];
  const float* qb = (const float*)d_in[6];
  const float* kW = (const float*)d_in[7];
  const float* kb = (const float*)d_in[8];
  const float* wH = (const float*)d_in[9];

  float* out  = (float*)d_out;               // [B,S,D]
  float* attn = out + (size_t)B_ * S_ * D_;  // [B,S,H,T]

  // workspace: bf16 qs/ks [B,H,S,DH], vs [B,DH,S], head [B,S,H,DH] (52 MB)
  __bf16* qs_ws = (__bf16*)d_ws;
  __bf16* ks_ws = qs_ws + (size_t)B_ * H_ * S_ * DH_;
  __bf16* vs_ws = ks_ws + (size_t)B_ * H_ * S_ * DH_;
  __bf16* hd_ws = vs_ws + (size_t)B_ * S_ * DH_;

  proj_kernel<<<dim3(8192 / 128, DH_ / 64, 9), 256, 0, stream>>>(
      q, k, v, qW, kW, vW, qb, kb, vb, qs_ws, ks_ws, vs_ws);

  const size_t attn_lds = (size_t)16 * 2048 * 4 +
                          ((size_t)16 * 2056 + 256 * 136 + 16 * 264) * 2 +
                          (256 + 16 + 16) * 4;
  attn_kernel<<<dim3(S_ / 16, H_, B_), 256, attn_lds, stream>>>(
      qs_ws, ks_ws, vs_ws, attn, hd_ws);

  const size_t out_lds = (size_t)(64 + 128) * 264 * 2;
  outproj_kernel<<<dim3(8192 / 64, D_ / 128), 256, out_lds, stream>>>(
      hd_ws, wH, out);
}
